// MultiHeadAttention_49100066128360
// MI455X (gfx1250) — matrix operand order repaired and checked by reading, not compiled, not run on hardware
//
#include <hip/hip_runtime.h>
#include <hip/hip_bf16.h>

// ---------------------------------------------------------------------------
// MHA forward for MI455X (gfx1250): bf16 WMMA pipeline.
//   B=2, S=2048, DIM=1024, HEADS=16, DK=64
// ---------------------------------------------------------------------------

#define B_SZ   2
#define SEQ    2048
#define SEQ_LG 11
#define DIM_   1024
#define HEADS_ 16
#define DK_    64

typedef __bf16 bf16;
typedef bf16  v16bf __attribute__((ext_vector_type(16)));
typedef bf16  v8bf  __attribute__((ext_vector_type(8)));
typedef float v8f   __attribute__((ext_vector_type(8)));

union V16U { v16bf v; v8bf h[2]; };

__device__ __forceinline__ bf16 f2bf(float f) {
    union { float f; unsigned u; } x; x.f = f;
    unsigned r = x.u + 0x7FFFu + ((x.u >> 16) & 1u);   // round-to-nearest-even
    unsigned short hs = (unsigned short)(r >> 16);
    bf16 out; __builtin_memcpy(&out, &hs, 2);
    return out;
}

// ---------------------------------------------------------------------------
// fp32 -> bf16 elementwise cast
// ---------------------------------------------------------------------------
__global__ void cast_f32_bf16(const float* __restrict__ in, bf16* __restrict__ out, int n) {
    int i = blockIdx.x * blockDim.x + threadIdx.x;
    if (i < n) out[i] = f2bf(in[i]);
}

// ---------------------------------------------------------------------------
// C[M,N] = A[M,K] @ W[N,K]^T  via v_wmma_f32_16x16x32_bf16
//   mode 0: store bf16 row-major [M,N]
//   mode 1: store bf16 transposed-per-batch for V: [(b*HEADS+h)*DK+d][Sdim]
//   mode 2: store f32 [M,N] + bias (final output projection)
// One wave computes a 32x64 tile (2 M-subtiles x 4 N-subtiles = 8 accumulators);
// B operands reused in-register across both M-subtiles. Block = 4 waves = 128x64.
// ---------------------------------------------------------------------------
__global__ __launch_bounds__(128)
void gemm_bf16_wmma(const bf16* __restrict__ A, const bf16* __restrict__ W,
                    bf16* __restrict__ Cb, float* __restrict__ Cf,
                    const float* __restrict__ bias,
                    int M, int N, int K, int mode, int sLog2)
{
    const int lane = threadIdx.x & 31;
    const int wave = threadIdx.x >> 5;
    const int half = lane >> 4;
    const int l16  = lane & 15;

    const int m0 = (blockIdx.x * 4 + wave) * 32;      // 32 rows per wave
    const int n0 = blockIdx.y * 64;

    const bf16* arow0 = A + (size_t)(m0 + l16) * K;        // A tile 0: rows m0..m0+15
    const bf16* arow1 = A + (size_t)(m0 + 16 + l16) * K;   // A tile 1: rows m0+16..m0+31
    const bf16* wrow  = W + (size_t)(n0 + l16) * K;        // B col n = l16 -> W row

    v8f acc[2][4] = {{{}, {}, {}, {}}, {{}, {}, {}, {}}};

    for (int k0 = 0; k0 < K; k0 += 32) {
        // A operands 16x32: 16 consecutive K values at k0 + 16*half
        V16U a0, a1;
        a0.h[0] = *(const v8bf*)(arow0 + k0 + half * 16);
        a0.h[1] = *(const v8bf*)(arow0 + k0 + half * 16 + 8);
        a1.h[0] = *(const v8bf*)(arow1 + k0 + half * 16);
        a1.h[1] = *(const v8bf*)(arow1 + k0 + half * 16 + 8);
        __builtin_prefetch(arow0 + k0 + 64, 0, 1);
        __builtin_prefetch(arow1 + k0 + 64, 0, 1);

        // B operands: 16 consecutive K values at k0 + 16*half, one tile per 16 cols
        const bf16* wk = wrow + k0 + half * 16;
        v16bf bm[4];
#pragma unroll
        for (int nt = 0; nt < 4; ++nt)
            bm[nt] = *(const v16bf*)(wk + (size_t)nt * 16 * K);

#pragma unroll
        for (int nt = 0; nt < 4; ++nt) {
            acc[0][nt] = __builtin_amdgcn_wmma_f32_16x16x32_bf16(
                false, a0.v, false, bm[nt], (short)0, acc[0][nt], false, false);
            acc[1][nt] = __builtin_amdgcn_wmma_f32_16x16x32_bf16(
                false, a1.v, false, bm[nt], (short)0, acc[1][nt], false, false);
        }
    }

    // D layout: lane holds N = l16 within tile, VGPR r holds row r + 8*half
#pragma unroll
    for (int mt = 0; mt < 2; ++mt) {
        const int mb = m0 + mt * 16;
        if (mode == 2) {
#pragma unroll
            for (int nt = 0; nt < 4; ++nt) {
                int col = n0 + nt * 16 + l16;
                float bs = bias ? bias[col] : 0.0f;
#pragma unroll
                for (int r = 0; r < 8; ++r) {
                    int row = mb + r + 8 * half;
                    Cf[(size_t)row * N + col] = acc[mt][nt][r] + bs;
                }
            }
        } else if (mode == 1) {
#pragma unroll
            for (int nt = 0; nt < 4; ++nt) {
                int col = n0 + nt * 16 + l16;                  // = h*DK + d
#pragma unroll
                for (int r = 0; r < 8; ++r) {
                    int row = mb + r + 8 * half;               // = b*S + t
                    int bb = row >> sLog2;
                    int t  = row - (bb << sLog2);
                    Cb[((size_t)bb * N + col) * (1 << sLog2) + t] = f2bf(acc[mt][nt][r]);
                }
            }
        } else {
#pragma unroll
            for (int nt = 0; nt < 4; ++nt) {
                int col = n0 + nt * 16 + l16;
#pragma unroll
                for (int r = 0; r < 8; ++r) {
                    int row = mb + r + 8 * half;
                    Cb[(size_t)row * N + col] = f2bf(acc[mt][nt][r]);
                }
            }
        }
    }
}

// ---------------------------------------------------------------------------
// Flash attention, causal. One wave per (b, h, 16-row query tile).
//   Qp, Kp : bf16 [B*S, DIM] (heads packed: col = h*DK + d)
//   Vt     : bf16 [(b*HEADS+h)*DK + d][SEQ]   (pre-transposed by GEMM mode 1)
//   Oa     : bf16 [B*S, DIM]
// ---------------------------------------------------------------------------
__global__ __launch_bounds__(32)
void flash_attn_causal(const bf16* __restrict__ Qp, const bf16* __restrict__ Kp,
                       const bf16* __restrict__ Vt, bf16* __restrict__ Oa)
{
    const int qt   = blockIdx.x;
    const int h    = blockIdx.y;
    const int b    = blockIdx.z;
    const int lane = threadIdx.x;
    const int half = lane >> 4;
    const int l16  = lane & 15;
    const int qrow0 = qt * 16;

    __shared__ __align__(16) bf16 pLds[16 * 40];   // P tile, pitch 40 (80B, 16B-mult)

    // --- Q A-operands: two 16x32 slabs covering dk = 0..63 ---
    const bf16* qbase = Qp + ((size_t)(b * SEQ + qrow0 + l16)) * DIM_ + h * DK_;
    V16U a0, a1;
    a0.h[0] = *(const v8bf*)(qbase + half * 16);
    a0.h[1] = *(const v8bf*)(qbase + half * 16 + 8);
    a1.h[0] = *(const v8bf*)(qbase + 32 + half * 16);
    a1.h[1] = *(const v8bf*)(qbase + 32 + half * 16 + 8);

    float mrow[8], lrow[8];
    v8f o[4] = {{}, {}, {}, {}};
#pragma unroll
    for (int r = 0; r < 8; ++r) { mrow[r] = -3.0e38f; lrow[r] = 0.0f; }

    const bf16* kbase = Kp + (size_t)(b * SEQ) * DIM_ + h * DK_;
    const bf16* vbase = Vt + (size_t)((b * HEADS_ + h) * DK_) * SEQ;

    const int nkb = (qrow0 + 16 + 31) / 32;        // 32-key blocks up to diagonal
    for (int kb = 0; kb < nkb; ++kb) {
        const int t0 = kb * 32;

        // --- scores S = Q(16x64) @ K^T : two 16x16 tiles, each K-dim 64 = 2 WMMAs
        const bf16* kr0 = kbase + (size_t)(t0 + l16) * DIM_ + half * 16;
        const bf16* kr1 = kbase + (size_t)(t0 + 16 + l16) * DIM_ + half * 16;
        v16bf bk0lo = *(const v16bf*)(kr0);
        v16bf bk0hi = *(const v16bf*)(kr0 + 32);
        v16bf bk1lo = *(const v16bf*)(kr1);
        v16bf bk1hi = *(const v16bf*)(kr1 + 32);

        v8f s0 = {}, s1 = {};
        s0 = __builtin_amdgcn_wmma_f32_16x16x32_bf16(false, a0.v, false, bk0lo, (short)0, s0, false, false);
        s0 = __builtin_amdgcn_wmma_f32_16x16x32_bf16(false, a1.v, false, bk0hi, (short)0, s0, false, false);
        s1 = __builtin_amdgcn_wmma_f32_16x16x32_bf16(false, a0.v, false, bk1lo, (short)0, s1, false, false);
        s1 = __builtin_amdgcn_wmma_f32_16x16x32_bf16(false, a1.v, false, bk1hi, (short)0, s1, false, false);

        // --- scale + causal mask (row = qrow0 + r + 8*half; key = t0 + nt*16 + l16)
        const int tA = t0 + l16, tB = t0 + 16 + l16;
        float p0[8], p1[8];
#pragma unroll
        for (int r = 0; r < 8; ++r) {
            int row = qrow0 + r + 8 * half;
            p0[r] = (tA <= row) ? s0[r] * 0.125f : -1.0e9f;
            p1[r] = (tB <= row) ? s1[r] * 0.125f : -1.0e9f;
        }

        // --- online softmax: row max / rescale / exp / row sum (16-lane halves)
        float mnew[8], alpha[8];
#pragma unroll
        for (int r = 0; r < 8; ++r) {
            float v = fmaxf(p0[r], p1[r]);
#pragma unroll
            for (int off = 1; off < 16; off <<= 1)
                v = fmaxf(v, __shfl_xor(v, off, 32));
            mnew[r]  = fmaxf(mrow[r], v);
            alpha[r] = __expf(mrow[r] - mnew[r]);
            mrow[r]  = mnew[r];
        }
#pragma unroll
        for (int r = 0; r < 8; ++r) {
            p0[r] = __expf(p0[r] - mnew[r]);
            p1[r] = __expf(p1[r] - mnew[r]);
            float s = p0[r] + p1[r];
#pragma unroll
            for (int off = 1; off < 16; off <<= 1)
                s += __shfl_xor(s, off, 32);
            lrow[r] = lrow[r] * alpha[r] + s;
            o[0][r] *= alpha[r]; o[1][r] *= alpha[r];
            o[2][r] *= alpha[r]; o[3][r] *= alpha[r];
        }

        // --- transpose P (16 rows x 32 keys) through LDS into A-operand layout
        __syncthreads();
#pragma unroll
        for (int r = 0; r < 8; ++r) {
            int m = r + 8 * half;
            pLds[m * 40 + l16]      = f2bf(p0[r]);
            pLds[m * 40 + 16 + l16] = f2bf(p1[r]);
        }
        __syncthreads();
        V16U pa;
        pa.h[0] = *(const v8bf*)(&pLds[l16 * 40 + half * 16]);
        pa.h[1] = *(const v8bf*)(&pLds[l16 * 40 + half * 16 + 8]);

        // --- O += P(16x32) @ V(32x64): V B-operand contiguous thanks to Vt layout
        const bf16* vr = vbase + (size_t)l16 * SEQ + t0 + half * 16;
#pragma unroll
        for (int nt = 0; nt < 4; ++nt) {
            v16bf bv = *(const v16bf*)(vr + (size_t)nt * 16 * SEQ);
            o[nt] = __builtin_amdgcn_wmma_f32_16x16x32_bf16(
                false, pa.v, false, bv, (short)0, o[nt], false, false);
        }
    }

    // --- epilogue: divide by l, write bf16 [b, row, h*DK + d]
    bf16* obase = Oa + ((size_t)(b * SEQ + qrow0)) * DIM_ + h * DK_;
#pragma unroll
    for (int r = 0; r < 8; ++r) {
        float inv = 1.0f / lrow[r];
        size_t ro = (size_t)(r + 8 * half) * DIM_;
#pragma unroll
        for (int nt = 0; nt < 4; ++nt)
            obase[ro + nt * 16 + l16] = f2bf(o[nt][r] * inv);
    }
}

// ---------------------------------------------------------------------------
extern "C" void kernel_launch(void* const* d_in, const int* in_sizes, int n_in,
                              void* d_out, int out_size, void* d_ws, size_t ws_size,
                              hipStream_t stream) {
    const float* Q   = (const float*)d_in[0];
    const float* K   = (const float*)d_in[1];
    const float* V   = (const float*)d_in[2];
    /* d_in[3] = mask: causal handled analytically */
    const float* wq  = (const float*)d_in[4];
    const float* wk  = (const float*)d_in[5];
    const float* wv  = (const float*)d_in[6];
    const float* wo  = (const float*)d_in[7];
    const float* wob = (const float*)d_in[8];
    float* out = (float*)d_out;

    const size_t nX = (size_t)B_SZ * SEQ * DIM_;   // 4,194,304
    const size_t nW = (size_t)DIM_ * DIM_;         // 1,048,576
    const int M = B_SZ * SEQ;                      // 4096

    char* ws = (char*)d_ws;
    bf16* Qb  = (bf16*)ws; ws += nX * 2;
    bf16* Kb  = (bf16*)ws; ws += nX * 2;
    bf16* Vb  = (bf16*)ws; ws += nX * 2;
    bf16* wqb = (bf16*)ws; ws += nW * 2;
    bf16* wkb = (bf16*)ws; ws += nW * 2;
    bf16* wvb = (bf16*)ws; ws += nW * 2;
    bf16* wob_= (bf16*)ws; ws += nW * 2;
    bf16* qp  = (bf16*)ws; ws += nX * 2;
    bf16* kp  = (bf16*)ws; ws += nX * 2;
    bf16* vt  = (bf16*)ws; ws += nX * 2;
    bf16* ao  = (bf16*)ws; ws += nX * 2;

    cast_f32_bf16<<<(int)((nX + 255) / 256), 256, 0, stream>>>(Q,  Qb,  (int)nX);
    cast_f32_bf16<<<(int)((nX + 255) / 256), 256, 0, stream>>>(K,  Kb,  (int)nX);
    cast_f32_bf16<<<(int)((nX + 255) / 256), 256, 0, stream>>>(V,  Vb,  (int)nX);
    cast_f32_bf16<<<(int)((nW + 255) / 256), 256, 0, stream>>>(wq, wqb, (int)nW);
    cast_f32_bf16<<<(int)((nW + 255) / 256), 256, 0, stream>>>(wk, wkb, (int)nW);
    cast_f32_bf16<<<(int)((nW + 255) / 256), 256, 0, stream>>>(wv, wvb, (int)nW);
    cast_f32_bf16<<<(int)((nW + 255) / 256), 256, 0, stream>>>(wo, wob_,(int)nW);

    dim3 gg(M / 128, DIM_ / 64);   // (32, 16): 4 waves/block, 32x64 tile/wave
    gemm_bf16_wmma<<<gg, 128, 0, stream>>>(Qb, wqb, qp, nullptr, nullptr, M, DIM_, DIM_, 0, SEQ_LG);
    gemm_bf16_wmma<<<gg, 128, 0, stream>>>(Kb, wkb, kp, nullptr, nullptr, M, DIM_, DIM_, 0, SEQ_LG);
    gemm_bf16_wmma<<<gg, 128, 0, stream>>>(Vb, wvb, vt, nullptr, nullptr, M, DIM_, DIM_, 1, SEQ_LG);

    dim3 fg(SEQ / 16, HEADS_, B_SZ);
    flash_attn_causal<<<fg, 32, 0, stream>>>(qp, kp, vt, ao);

    gemm_bf16_wmma<<<gg, 128, 0, stream>>>(ao, wob_, nullptr, out, wob, M, DIM_, DIM_, 2, SEQ_LG);
}
